// IALVQ_Pytorch_17600775979409
// MI455X (gfx1250) — matrix operand order repaired and checked by reading, not compiled, not run on hardware
//
#include <hip/hip_runtime.h>

typedef __attribute__((ext_vector_type(16))) _Float16 v16h;
typedef __attribute__((ext_vector_type(8)))  float    v8f;

#define NROWS 131072
#define FDIM  64
#define PROTO 512
#define NTILES (PROTO / 16)     // 32 prototype tiles
#define ROWS_PER_BLOCK 128      // 8 waves * 16 rows
#define THREADS 256
#define PLANE (PROTO * FDIM)    // halves per W plane (hi / lo)

// Setup: split prototypes into f16 hi/lo planes and compute ||w||^2.
__global__ void lvq_setup(const float* __restrict__ w,
                          _Float16* __restrict__ wh,   // [PROTO*FDIM] hi plane
                          _Float16* __restrict__ wl,   // [PROTO*FDIM] lo plane
                          float* __restrict__ w2) {
  int p = blockIdx.x * blockDim.x + threadIdx.x;
  if (p >= PROTO) return;
  const float* row = w + p * FDIM;
  float s = 0.f;
#pragma unroll
  for (int f = 0; f < FDIM; ++f) {
    float v = row[f];
    s += v * v;
    _Float16 h = (_Float16)v;
    wh[p * FDIM + f] = h;
    wl[p * FDIM + f] = (_Float16)(v - (float)h);
  }
  w2[p] = s;
}

// Build hi/lo 16x32 f16 A fragments (K-slab kb*32..+31) for this lane.
// Layout: half h -> K = h + 16*g + kb*32.
__device__ __forceinline__ void make_a2(const float4* xv, int kb, int g,
                                        v16h& ah, v16h& al) {
  int o = kb * 32 + 16 * g;
  float4 c0 = xv[o / 4],     c1 = xv[o / 4 + 1];
  float4 c2 = xv[o / 4 + 2], c3 = xv[o / 4 + 3];
  float v[16] = {c0.x, c0.y, c0.z, c0.w, c1.x, c1.y, c1.z, c1.w,
                 c2.x, c2.y, c2.z, c2.w, c3.x, c3.y, c3.z, c3.w};
#pragma unroll
  for (int i = 0; i < 16; ++i) {
    _Float16 h = (_Float16)v[i];
    ah[i] = h;
    al[i] = (_Float16)(v[i] - (float)h);
  }
}

__global__ void __launch_bounds__(THREADS)
lvq_predict(const float* __restrict__ x,
            const _Float16* __restrict__ wplanes,  // hi plane then lo plane
            const float* __restrict__ w2,
            const int* __restrict__ c_w,
            float* __restrict__ out) {
  extern __shared__ _Float16 Wl[];  // 2 * PROTO * FDIM halves = 128 KB

  // Cooperative copy of both W planes into LDS: 8192 uint4.
  {
    const uint4* src = (const uint4*)wplanes;
    uint4* dst = (uint4*)Wl;
#pragma unroll
    for (int i = 0; i < (2 * PLANE * 2 / 16) / THREADS; ++i)
      dst[threadIdx.x + THREADS * i] = src[threadIdx.x + THREADS * i];
  }
  __syncthreads();

  const int wave = threadIdx.x >> 5;
  const int lane = threadIdx.x & 31;
  const int m    = lane & 15;   // row-in-tile (A) / column (B,D)
  const int g    = lane >> 4;   // lane-half selector
  const int row0 = blockIdx.x * ROWS_PER_BLOCK + wave * 16;

  // Split-f16 A fragments: two K-slabs, hi and lo parts.
  const float4* xv = (const float4*)(x + (size_t)(row0 + m) * FDIM);
  v16h ah0, al0, ah1, al1;
  make_a2(xv, 0, g, ah0, al0);
  make_a2(xv, 1, g, ah1, al1);

  float best_val[8];
  int   best_idx[8];
#pragma unroll
  for (int r = 0; r < 8; ++r) { best_val[r] = 3.4e38f; best_idx[r] = 0; }

  // B fragments (hi/lo planes), double-buffered so next tile's LDS loads
  // overlap the current tile's 6-WMMA chain + argmin VALU.
  const _Float16* wbH = Wl + m * FDIM + 16 * g;
  const _Float16* wbL = wbH + PLANE;
  v16h bh0 = *(const v16h*)(wbH);
  v16h bh1 = *(const v16h*)(wbH + 32);
  v16h bl0 = *(const v16h*)(wbL);
  v16h bl1 = *(const v16h*)(wbL + 32);

#pragma unroll 2
  for (int t = 0; t < NTILES; ++t) {
    const int tn = (t + 1) & (NTILES - 1);        // wrap: harmless extra load
    const _Float16* wnH = wbH + tn * 16 * FDIM;
    const _Float16* wnL = wbL + tn * 16 * FDIM;
    v16h nh0 = *(const v16h*)(wnH);
    v16h nh1 = *(const v16h*)(wnH + 32);
    v16h nl0 = *(const v16h*)(wnL);
    v16h nl1 = *(const v16h*)(wnL + 32);

    // dot = xh*wh + xh*wl + xl*wh  (xl*wl term ~1e-5, dropped)
    v8f acc = {};
    acc = __builtin_amdgcn_wmma_f32_16x16x32_f16(false, ah0, false, bh0,
                                                 (short)0, acc, false, false);
    acc = __builtin_amdgcn_wmma_f32_16x16x32_f16(false, ah1, false, bh1,
                                                 (short)0, acc, false, false);
    acc = __builtin_amdgcn_wmma_f32_16x16x32_f16(false, ah0, false, bl0,
                                                 (short)0, acc, false, false);
    acc = __builtin_amdgcn_wmma_f32_16x16x32_f16(false, ah1, false, bl1,
                                                 (short)0, acc, false, false);
    acc = __builtin_amdgcn_wmma_f32_16x16x32_f16(false, al0, false, bh0,
                                                 (short)0, acc, false, false);
    acc = __builtin_amdgcn_wmma_f32_16x16x32_f16(false, al1, false, bh1,
                                                 (short)0, acc, false, false);

    const int   p   = t * 16 + m;
    const float w2v = w2[p];
#pragma unroll
    for (int r = 0; r < 8; ++r) {
      float s = fmaf(-2.0f, acc[r], w2v);
      if (s < best_val[r]) { best_val[r] = s; best_idx[r] = p; }
    }

    bh0 = nh0; bh1 = nh1; bl0 = nl0; bl1 = nl1;
  }

  // Reduce argmin across the 16 lanes of each half (D rows r+8g live there).
#pragma unroll
  for (int off = 1; off <= 8; off <<= 1) {
#pragma unroll
    for (int r = 0; r < 8; ++r) {
      float ov = __shfl_xor(best_val[r], off, 32);
      int   oi = __shfl_xor(best_idx[r], off, 32);
      if (ov < best_val[r] || (ov == best_val[r] && oi < best_idx[r])) {
        best_val[r] = ov; best_idx[r] = oi;
      }
    }
  }

  if (m == 0) {
#pragma unroll
    for (int r = 0; r < 8; ++r)
      out[row0 + 8 * g + r] = (float)c_w[best_idx[r]];
  }
}

extern "C" void kernel_launch(void* const* d_in, const int* in_sizes, int n_in,
                              void* d_out, int out_size, void* d_ws, size_t ws_size,
                              hipStream_t stream) {
  const float* x   = (const float*)d_in[0];
  // d_in[1] = y (unused by prediction)
  const float* w   = (const float*)d_in[2];
  const int*   c_w = (const int*)d_in[3];
  float*       out = (float*)d_out;

  _Float16* wh = (_Float16*)d_ws;                       // hi plane (64 KB)
  _Float16* wl = wh + PLANE;                            // lo plane (64 KB)
  float*    w2 = (float*)((char*)d_ws + 2 * PLANE * sizeof(_Float16));

  lvq_setup<<<(PROTO + 255) / 256, 256, 0, stream>>>(w, wh, wl, w2);
  lvq_predict<<<NROWS / ROWS_PER_BLOCK, THREADS,
                2 * PLANE * sizeof(_Float16), stream>>>(x, wh, w2, c_w, out);
}
